// Model_50783693307986
// MI455X (gfx1250) — matrix operand order repaired and checked by reading, not compiled, not run on hardware
//
#include <hip/hip_runtime.h>
#include <hip/hip_bf16.h>
#include <stdint.h>

// ---------------- WMMA types (gfx1250, wave32) ----------------
typedef __attribute__((ext_vector_type(16))) _Float16 v16h;
typedef __attribute__((ext_vector_type(8)))  float    v8f;

#define BB 128
#define TT 512
#define EE 128
#define HH 256
#define KK 32
#define FOURH 1024      // 4*H
#define KCAT 384        // E + H

// ---------------- fragment loaders ----------------
// A-matrix 16x32 f16: lane L holds row m=L&15; lanes 0-15 cover K 0-7 & 16-23,
// lanes 16-31 cover K 8-15 & 24-31 (ISA 7.12.2). Two 16B loads per lane.
__device__ inline v16h load_a_frag(const _Float16* p /* at (m, k0+kbase) */) {
    union { v16h v; uint4 u[2]; } r;
    r.u[0] = *reinterpret_cast<const uint4*>(p);
    r.u[1] = *reinterpret_cast<const uint4*>(p + 16);
    return r.v;
}
// B-matrix 32x16 f16: lane L holds column n=L&15; lanes 0-15 K 0-7 & 16-23,
// lanes 16-31 K 8-15 & 24-31 (same K order as A) from W[n][k] row.
__device__ inline v16h load_b_frag(const _Float16* p /* at (n, k0+koff) */) {
    union { v16h v; uint4 u[2]; } r;
    r.u[0] = *reinterpret_cast<const uint4*>(p);
    r.u[1] = *reinterpret_cast<const uint4*>(p + 16);
    return r.v;
}

__device__ inline float sigm(float x) { return 1.0f / (1.0f + __expf(-x)); }

// ---------------- prep kernels ----------------
// Wcat[n][k], n in [0,4H), k in [0,E+H): k<E -> W_ih[n][k], else W_hh[n][k-E]
__global__ __launch_bounds__(256) void prep_wcat(const float* __restrict__ wih,
                                                 const float* __restrict__ whh,
                                                 _Float16* __restrict__ out) {
    int i = blockIdx.x * blockDim.x + threadIdx.x;
    if (i >= FOURH * KCAT) return;
    int n = i / KCAT, k = i - n * KCAT;
    float v = (k < EE) ? wih[n * EE + k] : whh[n * HH + (k - EE)];
    out[i] = (_Float16)v;
}

__global__ __launch_bounds__(256) void cvt_f16(const float* __restrict__ in,
                                               _Float16* __restrict__ out, int n) {
    int i = blockIdx.x * blockDim.x + threadIdx.x;
    if (i < n) out[i] = (_Float16)in[i];
}

// embedding gather -> f16, PAD(=0) row forced to zero
__global__ __launch_bounds__(256) void embed_kernel(const int* __restrict__ idx,
                                                    const float* __restrict__ table,
                                                    _Float16* __restrict__ x) {
    int i = blockIdx.x * blockDim.x + threadIdx.x;   // over B*T*E
    if (i >= BB * TT * EE) return;
    int bt = i >> 7, e = i & (EE - 1);
    int id = idx[bt];
    float v = (id == 0) ? 0.0f : table[(size_t)id * EE + e];
    x[i] = (_Float16)v;
}

// ---------------- fused bidirectional LSTM ----------------
// grid = (8 batch tiles, 2 directions), block = 512 threads (16 waves).
// Each block owns 16 batch rows and loops over all T steps; h_prev / c live in LDS.
// Per step: g[16,1024] = [x_t | h_prev] (16x384) x Wcat^T via WMMA, +bias,
// then gate nonlinearity, c/h update, h written to h_all[B,T,2H] (f16).
// Weight fragments are re-read from L2 each step (weights = 2x768KB, fully
// L2-resident). An opaque zero OFFSET (not an opaque pointer!) defeats LICM so
// the 48 B-fragments are not hoisted+spilled, while keeping pointer provenance
// so address-space inference still emits global_load (not flat_load, which
// would couple LOADcnt with DScnt and occupy the LDS datapath).
__global__ __launch_bounds__(512) void lstm_kernel(const _Float16* __restrict__ x,   // [B,T,E]
                                                   const _Float16* __restrict__ Wf,  // [4H,KCAT]
                                                   const _Float16* __restrict__ Wb,
                                                   const float* __restrict__ bias_f, // [4H]
                                                   const float* __restrict__ bias_b,
                                                   _Float16* __restrict__ h_all) {   // [B,T,2H]
    const int btile = blockIdx.x;        // 0..7
    const int dir   = blockIdx.y;        // 0 fwd, 1 bwd
    const _Float16* W    = dir ? Wb : Wf;
    const float*    bias = dir ? bias_b : bias_f;

    __shared__ _Float16 h_prev[16 * HH];     // 8 KB
    __shared__ float    c_st[16 * HH];       // 16 KB
    __shared__ float    g_lds[16 * FOURH];   // 64 KB

    const int tid  = threadIdx.x;
    const int wave = tid >> 5;
    const int lane = tid & 31;
    const int row0 = btile * 16;

    for (int i = tid; i < 16 * HH; i += 512) { h_prev[i] = (_Float16)0.0f; c_st[i] = 0.0f; }
    __syncthreads();

    const int m     = lane & 15;
    const int kbase = (lane >> 4) << 3;   // A-frag: 0 or 8
    const int koff  = (lane >> 4) << 3;   // B-frag: 0 or 8
    const int mhi   = (lane >> 4) << 3;   // C-frag row offset

    for (int step = 0; step < TT; ++step) {
        const int t = dir ? (TT - 1 - step) : step;

        // Opaque zero element-offset, fresh each iteration: blocks LICM on the
        // weight loads while preserving global address-space provenance.
        size_t wofs = 0;
        asm volatile("" : "+s"(wofs));
        const _Float16* Wt = W + wofs;

        // prefetch next step's x rows (global_prefetch_b8)
        if (step + 1 < TT) {
            const int tn = dir ? (TT - 2 - step) : (step + 1);
            __builtin_prefetch(&x[((size_t)(row0 + m) * TT + tn) * EE], 0, 0);
        }

        v8f acc[4];
#pragma unroll
        for (int nt = 0; nt < 4; ++nt) acc[nt] = (v8f)0.0f;

#pragma unroll
        for (int k0 = 0; k0 < KCAT; k0 += 32) {
            v16h af;
            if (k0 < EE) {
                af = load_a_frag(&x[((size_t)(row0 + m) * TT + t) * EE + k0 + kbase]);
            } else {
                af = load_a_frag(&h_prev[m * HH + (k0 - EE) + kbase]);
            }
#pragma unroll
            for (int nt = 0; nt < 4; ++nt) {
                const int n = (wave * 4 + nt) * 16 + (lane & 15);
                v16h bf = load_b_frag(&Wt[(size_t)n * KCAT + k0 + koff]);
                acc[nt] = __builtin_amdgcn_wmma_f32_16x16x32_f16(
                    false, af, false, bf, (short)0, acc[nt], false, false);
            }
        }

        // accumulators -> LDS with bias (C layout: lane = col, vgpr r = row)
#pragma unroll
        for (int nt = 0; nt < 4; ++nt) {
            const int n = (wave * 4 + nt) * 16 + (lane & 15);
            const float bv = bias[n];
#pragma unroll
            for (int r = 0; r < 8; ++r)
                g_lds[(mhi + r) * FOURH + n] = acc[nt][r] + bv;
        }
        __syncthreads();

        // gates: i,f,g,o at columns [0,H),[H,2H),[2H,3H),[3H,4H)
        for (int i = tid; i < 16 * HH; i += 512) {
            const int mm = i >> 8;          // row
            const int hc = i & (HH - 1);    // hidden col
            const float gi = g_lds[mm * FOURH + hc];
            const float gf = g_lds[mm * FOURH + HH + hc];
            const float gg = g_lds[mm * FOURH + 2 * HH + hc];
            const float go = g_lds[mm * FOURH + 3 * HH + hc];
            const float cn = sigm(gf) * c_st[i] + sigm(gi) * tanhf(gg);
            const float hn = sigm(go) * tanhf(cn);
            c_st[i] = cn;
            const _Float16 hh = (_Float16)hn;
            h_prev[i] = hh;
            h_all[((size_t)(row0 + mm) * TT + t) * (2 * HH) + dir * HH + hc] = hh;
        }
        __syncthreads();
    }
}

// ---------------- emissions GEMM: em[B*T,32] = h_all(f16) x linw^T + b ----------------
__global__ __launch_bounds__(256) void emis_kernel(const _Float16* __restrict__ h_all, // [B*T, 2H]
                                                   const _Float16* __restrict__ linw,  // [K, 2H]
                                                   const float* __restrict__ linb,     // [K]
                                                   float* __restrict__ em) {           // [B*T, K]
    const int wave = threadIdx.x >> 5;
    const int lane = threadIdx.x & 31;
    const int row0 = (blockIdx.x * 8 + wave) * 16;
    const int m     = lane & 15;
    const int kbase = (lane >> 4) << 3;
    const int koff  = (lane >> 4) << 3;
    const int mhi   = (lane >> 4) << 3;

    v8f acc0 = (v8f)0.0f, acc1 = (v8f)0.0f;
#pragma unroll
    for (int k0 = 0; k0 < 2 * HH; k0 += 32) {
        v16h af = load_a_frag(&h_all[(size_t)(row0 + m) * (2 * HH) + k0 + kbase]);
        v16h b0 = load_b_frag(&linw[(size_t)((lane & 15)) * (2 * HH) + k0 + koff]);
        v16h b1 = load_b_frag(&linw[(size_t)(16 + (lane & 15)) * (2 * HH) + k0 + koff]);
        acc0 = __builtin_amdgcn_wmma_f32_16x16x32_f16(false, af, false, b0, (short)0, acc0, false, false);
        acc1 = __builtin_amdgcn_wmma_f32_16x16x32_f16(false, af, false, b1, (short)0, acc1, false, false);
    }
    const int n0 = (lane & 15), n1 = 16 + (lane & 15);
    const float bb0 = linb[n0], bb1 = linb[n1];
#pragma unroll
    for (int r = 0; r < 8; ++r) {
        em[(size_t)(row0 + mhi + r) * KK + n0] = acc0[r] + bb0;
        em[(size_t)(row0 + mhi + r) * KK + n1] = acc1[r] + bb1;
    }
}

// ---------------- Viterbi: one wave32 per batch row (lane = tag), fused backtrace ----------------
__global__ __launch_bounds__(128) void viterbi_kernel(const float* __restrict__ em,     // [B,T,K]
                                                      const unsigned char* __restrict__ mask, // [B,T]
                                                      const float* __restrict__ start_t,
                                                      const float* __restrict__ end_t,
                                                      const float* __restrict__ trans,  // [K,K]
                                                      unsigned char* __restrict__ hist, // [T-1,B,K]
                                                      int* __restrict__ tags) {         // [B,T]
    __shared__ float tr[KK * KK];
    const int tid = threadIdx.x;
    for (int i = tid; i < KK * KK; i += 128) tr[i] = trans[i];
    __syncthreads();

    const int lane = tid & 31;
    const int b = blockIdx.x * 4 + (tid >> 5);

    float score = start_t[lane] + em[(size_t)b * TT * KK + lane];
    for (int t = 1; t < TT; ++t) {
        const float e = em[((size_t)b * TT + t) * KK + lane];
        float best = -3.4e38f;
        int bi = 0;
#pragma unroll 8
        for (int k1 = 0; k1 < KK; ++k1) {
            const float cand = __shfl(score, k1, 32) + tr[k1 * KK + lane];
            if (cand > best) { best = cand; bi = k1; }
        }
        hist[((size_t)(t - 1) * BB + b) * KK + lane] = (unsigned char)bi;
        const bool mk = mask[b * TT + t] != 0;
        score = mk ? (best + e) : score;
    }
    score += end_t[lane];

    // argmax across lanes (lowest index on ties, matching jnp.argmax)
    float s = score; int bestk = lane;
    for (int off = 16; off > 0; off >>= 1) {
        const float so = __shfl_down(s, off, 32);
        const int   ko = __shfl_down(bestk, off, 32);
        if (so > s || (so == s && ko < bestk)) { s = so; bestk = ko; }
    }
    bestk = __shfl(bestk, 0, 32);

    if (lane == 0) {
        int cur = bestk;
        tags[b * TT + (TT - 1)] = cur;
        for (int t = TT - 2; t >= 0; --t) {
            const bool mk = mask[b * TT + t + 1] != 0;
            const int idx = hist[((size_t)t * BB + b) * KK + cur];
            cur = mk ? idx : cur;
            tags[b * TT + t] = cur;
        }
    }
}

// ---------------- host side ----------------
extern "C" void kernel_launch(void* const* d_in, const int* in_sizes, int n_in,
                              void* d_out, int out_size, void* d_ws, size_t ws_size,
                              hipStream_t stream) {
    (void)in_sizes; (void)n_in; (void)out_size; (void)ws_size;

    const int*           inp     = (const int*)d_in[0];
    const unsigned char* mask    = (const unsigned char*)d_in[1];
    /* d_in[2] target: unused in decode */
    const float* table   = (const float*)d_in[3];
    const float* wih_f   = (const float*)d_in[4];
    const float* whh_f   = (const float*)d_in[5];
    const float* b_f     = (const float*)d_in[6];
    const float* wih_b   = (const float*)d_in[7];
    const float* whh_b   = (const float*)d_in[8];
    const float* b_b     = (const float*)d_in[9];
    const float* linw    = (const float*)d_in[10];
    const float* linb    = (const float*)d_in[11];
    const float* start_t = (const float*)d_in[12];
    const float* end_t   = (const float*)d_in[13];
    const float* trans   = (const float*)d_in[14];
    int* tags = (int*)d_out;

    // workspace carve (256B aligned)
    char* ws = (char*)d_ws;
    size_t off = 0;
    auto carve = [&](size_t bytes) { char* p = ws + off; off = (off + bytes + 255) & ~(size_t)255; return p; };
    _Float16* x_f16  = (_Float16*)carve((size_t)BB * TT * EE * 2);       // 16.8 MB
    _Float16* Wf     = (_Float16*)carve((size_t)FOURH * KCAT * 2);       // 768 KB
    _Float16* Wb     = (_Float16*)carve((size_t)FOURH * KCAT * 2);
    _Float16* lw16   = (_Float16*)carve((size_t)KK * 2 * HH * 2);        // 32 KB
    _Float16* h_all  = (_Float16*)carve((size_t)BB * TT * 2 * HH * 2);   // 67 MB
    float*    em     = (float*)carve((size_t)BB * TT * KK * 4);          // 8.4 MB
    unsigned char* hist = (unsigned char*)carve((size_t)(TT - 1) * BB * KK); // 2 MB

    // 1) weight prep
    prep_wcat<<<(FOURH * KCAT + 255) / 256, 256, 0, stream>>>(wih_f, whh_f, Wf);
    prep_wcat<<<(FOURH * KCAT + 255) / 256, 256, 0, stream>>>(wih_b, whh_b, Wb);
    cvt_f16<<<(KK * 2 * HH + 255) / 256, 256, 0, stream>>>(linw, lw16, KK * 2 * HH);

    // 2) embedding gather
    embed_kernel<<<(BB * TT * EE) / 256, 256, 0, stream>>>(inp, table, x_f16);

    // 3) fused bidirectional LSTM (single launch; each block loops over T)
    lstm_kernel<<<dim3(8, 2), 512, 0, stream>>>(x_f16, Wf, Wb, b_f, b_b, h_all);

    // 4) emissions GEMM
    emis_kernel<<<(BB * TT) / (16 * 8), 256, 0, stream>>>(h_all, lw16, linb, em);

    // 5) Viterbi decode + backtrace
    viterbi_kernel<<<BB / 4, 128, 0, stream>>>(em, mask, start_t, end_t, trans, hist, tags);
}
